// Net_12446815224381
// MI455X (gfx1250) — matrix operand order repaired and checked by reading, not compiled, not run on hardware
//
#include <hip/hip_runtime.h>
#include <hip/hip_bf16.h>
#include <math.h>

typedef _Float16 f16;
typedef __attribute__((ext_vector_type(8)))  _Float16 v8h;
typedef __attribute__((ext_vector_type(16))) _Float16 v16h;
typedef __attribute__((ext_vector_type(8)))  float    v8f;

#define BATCH 8
#define NPTS  4096
#define FIN   8
#define M1C   819
#define M2C   204
#define KNBR  64
#define BN_SCALE 0.9999950000374997f

// ---------------------------------------------------------------------------
// One 16x16x32 f16 WMMA step.
// A (16x32 f16): lane L row (L&15); half=(L>>4): vector elems 0-7 = K
//   half*8..half*8+7, elems 8-15 = 16+half*8..  -> two contiguous 16B runs.
// B: weights are PRE-SWIZZLED fragment-major: per (N-tile,K-tile), lane L's
//   16 halves live at tile*512 + L*16 -> one contiguous 32B run per lane.
// C/D (16x16 f32): lane L col (L&15); VGPR v row = v + half*8.
// ---------------------------------------------------------------------------
__device__ __forceinline__ v8f wmma_step(v8f acc, const f16* __restrict__ ap,
                                         const f16* __restrict__ bp)
{
  v8h alo = *(const v8h*)ap;          // 16B
  v8h ahi = *(const v8h*)(ap + 16);   // 16B
  v16h a = __builtin_shufflevector(alo, ahi, 0, 1, 2, 3, 4, 5, 6, 7,
                                   8, 9, 10, 11, 12, 13, 14, 15);
  v16h b = *(const v16h*)bp;          // 32B contiguous
  return __builtin_amdgcn_wmma_f32_16x16x32_f16(false, a, false, b, (short)0,
                                                acc, false, false);
}

// ---------------------------------------------------------------------------
// Generic GEMM: Out[rows,Cout] = act(A[rows,K] @ W + bias), W fragment-major.
// act: 0=none, 1=relu, 2=relu*BN_SCALE.  K % 32 == 0, Cout % 16 == 0.
// One 16x16 tile per wave, 8 waves per block.
// ---------------------------------------------------------------------------
__global__ void __launch_bounds__(256)
gemm_f16(const f16* __restrict__ A, const f16* __restrict__ Wf,
         const float* __restrict__ bias, f16* __restrict__ Out,
         int rows, int K, int Cout, int act)
{
  int lane = threadIdx.x & 31;
  int wv   = blockIdx.x * (blockDim.x >> 5) + (threadIdx.x >> 5);
  int ntN  = Cout >> 4;
  int ntM  = (rows + 15) >> 4;
  if (wv >= ntM * ntN) return;
  int mi = (wv / ntN) << 4;
  int ni = (wv % ntN) << 4;
  int l15 = lane & 15, half = lane >> 4;
  int arow  = mi + l15;
  int arowC = arow < rows ? arow : rows - 1;   // clamp; stores are guarded
  int nKT = K >> 5;
  const f16* ap = A + (size_t)arowC * K + half * 8;
  const f16* bp = Wf + ((size_t)(ni >> 4) * nKT) * 512 + lane * 16;
  v8f acc = {};
  for (int kt = 0; kt < nKT; ++kt)
    acc = wmma_step(acc, ap + (kt << 5), bp + (size_t)kt * 512);
  float bv = bias[ni + l15];
#pragma unroll
  for (int v = 0; v < 8; ++v) {
    int row = mi + v + half * 8;
    if (row < rows) {
      float r = acc[v] + bv;
      if (act) { r = r > 0.f ? r : 0.f; if (act == 2) r *= BN_SCALE; }
      Out[(size_t)row * Cout + ni + l15] = (f16)r;
    }
  }
}

// ---------------------------------------------------------------------------
// Fused SA (PointConv) module: gather [x_j, p_j - p_i] for QPB queries x 64
// neighbors, run 3-layer MLP entirely in LDS (ping-pong), radius-masked max.
// Weights are fragment-major both in global and in their LDS stage.
// ---------------------------------------------------------------------------
template <int QPB, int MAXC, int MAXW>
__global__ void __launch_bounds__(256)
sa_fused(const f16* __restrict__ Xin, int Cin, int n,
         const float* __restrict__ posS, const float* __restrict__ posQ,
         const int* __restrict__ nidx, const float* __restrict__ nd2,
         float r2, int m,
         const f16* __restrict__ W1, const float* __restrict__ b1, int CinPad, int C1,
         const f16* __restrict__ W2, const float* __restrict__ b2, int C2,
         const f16* __restrict__ W3, const float* __restrict__ b3, int C3,
         f16* __restrict__ Xout)
{
  constexpr int ROWS = QPB * 64;
  extern __shared__ char smem_raw[];
  f16* actA = (f16*)smem_raw;
  f16* actB = actA + ROWS * MAXC;
  f16* wsh  = actB + ROWS * MAXC;

  int tid = threadIdx.x;
  int qBlocks = (m + QPB - 1) / QPB;
  int b  = blockIdx.x / qBlocks;
  int q0 = (blockIdx.x % qBlocks) * QPB;

  // ---- build input features into actA (zero-padded to CinPad) ----
  for (int t = tid; t < ROWS * CinPad; t += blockDim.x) {
    int row = t / CinPad, c = t % CinPad;
    int qi = q0 + (row >> 6), k = row & 63;
    f16 v = (f16)0.f;
    if (qi < m) {
      int j = nidx[((size_t)b * m + qi) * KNBR + k];
      if (c < Cin) {
        v = Xin[((size_t)b * n + j) * Cin + c];
      } else if (c < Cin + 3) {
        int d = c - Cin;
        v = (f16)(posS[((size_t)b * n + j) * 3 + d] -
                  posQ[((size_t)b * m + qi) * 3 + d]);
      }
    }
    actA[row * MAXC + c] = v;
  }
  __syncthreads();

  const f16*   Ws[3] = {W1, W2, W3};
  const float* Bs[3] = {b1, b2, b3};
  int Kd[3] = {CinPad, C1, C2};
  int Co[3] = {C1, C2, C3};

  f16* src = actA;
  f16* dst = actB;
  int lane = tid & 31, l15 = lane & 15, half = lane >> 4;
  int wid = tid >> 5, nw = blockDim.x >> 5;

  for (int l = 0; l < 3; ++l) {
    int Kdim = Kd[l], Cout = Co[l];
    int nKT = Kdim >> 5;
    for (int t = tid; t < Kdim * Cout; t += blockDim.x) wsh[t] = Ws[l][t];
    __syncthreads();
    int ntN = Cout >> 4, ntM = ROWS >> 4;
    for (int t = wid; t < ntM * ntN; t += nw) {
      int mi = (t / ntN) << 4, ni = (t % ntN) << 4;
      const f16* ap = src + (mi + l15) * MAXC + half * 8;
      const f16* bp = wsh + ((ni >> 4) * nKT) * 512 + lane * 16;
      v8f acc = {};
      for (int kt = 0; kt < nKT; ++kt)
        acc = wmma_step(acc, ap + (kt << 5), bp + kt * 512);
      float bv = Bs[l][ni + l15];
#pragma unroll
      for (int v = 0; v < 8; ++v) {
        float r = acc[v] + bv;
        r = r > 0.f ? r : 0.f;
        r *= BN_SCALE;
        dst[(mi + v + half * 8) * MAXC + ni + l15] = (f16)r;
      }
    }
    __syncthreads();
    f16* tmp = src; src = dst; dst = tmp;
  }

  // ---- radius-masked max over the 64 neighbors of each query ----
  for (int t = tid; t < QPB * C3; t += blockDim.x) {
    int ql = t / C3, c = t % C3;
    int qi = q0 + ql;
    if (qi < m) {
      float mx = -INFINITY;
      for (int k = 0; k < KNBR; ++k) {
        float d = nd2[((size_t)b * m + qi) * KNBR + k];
        if (d <= r2) {
          float hv = (float)src[((ql << 6) + k) * MAXC + c];
          mx = hv > mx ? hv : mx;
        }
      }
      Xout[((size_t)b * m + qi) * C3 + c] = (f16)mx;
    }
  }
}

// ---------------------------------------------------------------------------
// Farthest point sampling: one block per batch, min-dist array in LDS.
// ---------------------------------------------------------------------------
__global__ void __launch_bounds__(256)
fps_kernel(const float* __restrict__ pos, int n, int m, int* __restrict__ idxOut)
{
  __shared__ float mind[4096];
  __shared__ float rv[256];
  __shared__ int   ri[256];
  __shared__ int   curS;
  __shared__ float cpt[3];
  int b = blockIdx.x, tid = threadIdx.x;
  const float* P = pos + (size_t)b * n * 3;
  for (int j = tid; j < n; j += blockDim.x) mind[j] = 3.4e38f;
  if (tid == 0) curS = 0;
  __syncthreads();
  for (int i = 0; i < m; ++i) {
    int cur = curS;
    if (tid == 0) idxOut[b * m + i] = cur;
    if (tid < 3) cpt[tid] = P[cur * 3 + tid];
    __syncthreads();
    float cx = cpt[0], cy = cpt[1], cz = cpt[2];
    float bv = -1.f; int bi = 0;
    for (int j = tid; j < n; j += blockDim.x) {
      float dx = P[j * 3] - cx, dy = P[j * 3 + 1] - cy, dz = P[j * 3 + 2] - cz;
      float d = dx * dx + dy * dy + dz * dz;
      float mm = mind[j]; mm = mm < d ? mm : d; mind[j] = mm;
      if (mm > bv) { bv = mm; bi = j; }
    }
    rv[tid] = bv; ri[tid] = bi;
    __syncthreads();
    for (int s = blockDim.x >> 1; s > 0; s >>= 1) {
      if (tid < s && rv[tid + s] > rv[tid]) { rv[tid] = rv[tid + s]; ri[tid] = ri[tid + s]; }
      __syncthreads();
    }
    if (tid == 0) curS = ri[0];
    __syncthreads();
  }
}

// ---------------------------------------------------------------------------
// K nearest neighbors: one block per (batch,query); all n distances in LDS,
// K iterative argmin reductions (matches lax.top_k(-d2, K) semantics).
// ---------------------------------------------------------------------------
__global__ void __launch_bounds__(256)
knn_kernel(const float* __restrict__ posQ, const float* __restrict__ posS,
           int m, int n, int K, int* __restrict__ nidx, float* __restrict__ nd2)
{
  __shared__ float d[4096];
  __shared__ float rv[256];
  __shared__ int   ri[256];
  int bq = blockIdx.x, b = bq / m, tid = threadIdx.x;
  float qx = posQ[(size_t)bq * 3], qy = posQ[(size_t)bq * 3 + 1], qz = posQ[(size_t)bq * 3 + 2];
  const float* P = posS + (size_t)b * n * 3;
  for (int j = tid; j < n; j += blockDim.x) {
    float dx = P[j * 3] - qx, dy = P[j * 3 + 1] - qy, dz = P[j * 3 + 2] - qz;
    d[j] = dx * dx + dy * dy + dz * dz;
  }
  __syncthreads();
  for (int k = 0; k < K; ++k) {
    float bv = 3.4e38f; int bi = 0;
    for (int j = tid; j < n; j += blockDim.x) {
      float v = d[j];
      if (v < bv) { bv = v; bi = j; }
    }
    rv[tid] = bv; ri[tid] = bi;
    __syncthreads();
    for (int s = blockDim.x >> 1; s > 0; s >>= 1) {
      if (tid < s && rv[tid + s] < rv[tid]) { rv[tid] = rv[tid + s]; ri[tid] = ri[tid + s]; }
      __syncthreads();
    }
    if (tid == 0) {
      nidx[(size_t)bq * K + k] = ri[0];
      nd2[(size_t)bq * K + k]  = rv[0];
      d[ri[0]] = 3.4e38f;
    }
    __syncthreads();
  }
}

// ---------------------------------------------------------------------------
// Weight pack: f32 [Cin,Cout] -> f16 fragment-major, K zero-padded to CinPad.
// dst[((niT*nKT + kkT)*32 + lane)*16 + j] = W[kkT*32 + (lane>>4)*8 + (j&7) + (j>>3)*16,
//                                             niT*16 + (lane&15)]
// ---------------------------------------------------------------------------
__global__ void pack_w_frag(const float* __restrict__ src, int Cin, int Cout,
                            int CinPad, f16* __restrict__ dst)
{
  int t = blockIdx.x * blockDim.x + threadIdx.x;
  if (t >= CinPad * Cout) return;
  int j    = t & 15;
  int lane = (t >> 4) & 31;
  int tile = t >> 9;
  int nKT  = CinPad >> 5;
  int kkT  = tile % nKT;
  int niT  = tile / nKT;
  int r = (kkT << 5) + ((lane >> 4) << 3) + (j & 7) + ((j >> 3) << 4);
  int c = (niT << 4) + (lane & 15);
  dst[t] = (r < Cin) ? (f16)src[(size_t)r * Cout + c] : (f16)0.f;
}

// ---------------------------------------------------------------------------
// Small utility kernels
// ---------------------------------------------------------------------------
__global__ void fill_f16(f16* p, int n)
{
  int t = blockIdx.x * blockDim.x + threadIdx.x;
  if (t < n) p[t] = (f16)0.f;
}

__global__ void copy_f16_mat(const f16* __restrict__ src, int rows, int C,
                             f16* __restrict__ dst, int stride, int colOff)
{
  int t = blockIdx.x * blockDim.x + threadIdx.x;
  if (t >= rows * C) return;
  int r = t / C, c = t % C;
  dst[(size_t)r * stride + colOff + c] = src[t];
}

__global__ void copy_f32_mat(const float* __restrict__ src, int rows, int C,
                             f16* __restrict__ dst, int stride, int colOff)
{
  int t = blockIdx.x * blockDim.x + threadIdx.x;
  if (t >= rows * C) return;
  int r = t / C, c = t % C;
  dst[(size_t)r * stride + colOff + c] = (f16)src[t];
}

__global__ void gather_pos(const float* __restrict__ pos, const int* __restrict__ idx,
                           int n, int m, float* __restrict__ out)
{
  int t = blockIdx.x * blockDim.x + threadIdx.x;
  if (t >= BATCH * m * 3) return;
  int c = t % 3, q = (t / 3) % m, b = t / (3 * m);
  out[t] = pos[((size_t)b * n + idx[b * m + q]) * 3 + c];
}

__global__ void rowmax_kernel(const f16* __restrict__ in, int m, int C,
                              f16* __restrict__ out)
{
  int t = blockIdx.x * blockDim.x + threadIdx.x;
  if (t >= BATCH * C) return;
  int c = t % C, b = t / C;
  float mx = -INFINITY;
  for (int i = 0; i < m; ++i) {
    float v = (float)in[((size_t)b * m + i) * C + c];
    mx = v > mx ? v : mx;
  }
  out[t] = (f16)mx;
}

__global__ void bcast_x3(const f16* __restrict__ x3, f16* __restrict__ dst,
                         int rowsPerB, int stride)
{
  int t = blockIdx.x * blockDim.x + threadIdx.x;
  if (t >= BATCH * rowsPerB * 1024) return;
  int c = t % 1024, r = t / 1024, b = r / rowsPerB;
  dst[(size_t)r * stride + c] = x3[b * 1024 + c];
}

// inverse-squared-distance weighted kNN interpolation (k<=3)
__global__ void interp_kernel(const f16* __restrict__ xc, const int* __restrict__ nidx,
                              const float* __restrict__ nd2, int nf, int nc, int C,
                              int Kk, f16* __restrict__ dst, int stride, int colOff)
{
  int t = blockIdx.x * blockDim.x + threadIdx.x;
  if (t >= BATCH * nf * C) return;
  int c = t % C, q = (t / C) % nf, b = t / (C * nf);
  float num = 0.f, den = 0.f;
  for (int k = 0; k < Kk; ++k) {
    float d = nd2[((size_t)b * nf + q) * Kk + k];
    float w = 1.f / fmaxf(d, 1e-16f);
    int j = nidx[((size_t)b * nf + q) * Kk + k];
    num += w * (float)xc[((size_t)b * nc + j) * C + c];
    den += w;
  }
  dst[((size_t)b * nf + q) * stride + colOff + c] = (f16)(num / den);
}

// final 64->8 linear + log_softmax (scalar; 8 outputs is below WMMA tile size)
__global__ void head_final(const f16* __restrict__ y, const float* __restrict__ W3,
                           const float* __restrict__ b3, float* __restrict__ out,
                           int rows)
{
  int r = blockIdx.x * blockDim.x + threadIdx.x;
  if (r >= rows) return;
  float acc[8];
#pragma unroll
  for (int c = 0; c < 8; ++c) acc[c] = b3[c];
  for (int k = 0; k < 64; ++k) {
    float xv = (float)y[(size_t)r * 64 + k];
#pragma unroll
    for (int c = 0; c < 8; ++c) acc[c] += xv * W3[k * 8 + c];
  }
  float mx = acc[0];
#pragma unroll
  for (int c = 1; c < 8; ++c) mx = acc[c] > mx ? acc[c] : mx;
  float s = 0.f;
#pragma unroll
  for (int c = 0; c < 8; ++c) s += expf(acc[c] - mx);
  float lse = mx + logf(s);
#pragma unroll
  for (int c = 0; c < 8; ++c) out[(size_t)r * 8 + c] = acc[c] - lse;
}

// ---------------------------------------------------------------------------
// Host orchestration
// ---------------------------------------------------------------------------
struct WInfo { int pidx, cin, cout, cinpad; };
static const WInfo wlist[18] = {
  {0, 11, 64, 32},   {2, 64, 64, 64},    {4, 64, 128, 64},     // sa1
  {6, 131, 128, 160},{8, 128, 128, 128}, {10, 128, 256, 128},  // sa2
  {12, 259, 256, 288},{14, 256, 512, 256},{16, 512, 1024, 512},// sa3
  {18, 1280, 256, 1280},{20, 256, 256, 256},                   // fp3
  {22, 384, 256, 384},{24, 256, 128, 256},                     // fp2
  {26, 136, 128, 160},{28, 128, 128, 128},{30, 128, 128, 128}, // fp1
  {32, 128, 128, 128},{34, 128, 64, 128},                      // head (L1,L2)
};

extern "C" void kernel_launch(void* const* d_in, const int* in_sizes, int n_in,
                              void* d_out, int out_size, void* d_ws, size_t ws_size,
                              hipStream_t stream)
{
  (void)in_sizes; (void)n_in; (void)out_size; (void)ws_size;
  const float* x   = (const float*)d_in[0];
  const float* pos = (const float*)d_in[1];
  const float* P[38];
  for (int i = 0; i < 38; ++i) P[i] = (const float*)d_in[2 + i];

  char* base = (char*)d_ws;
  size_t off = 0;
  auto alloc = [&](size_t bytes) -> char* {
    size_t o = (off + 255) & ~(size_t)255;
    off = o + bytes;
    return base + o;
  };

  f16*   xh    = (f16*)  alloc((size_t)BATCH * NPTS * FIN * 2);
  int*   idx1  = (int*)  alloc((size_t)BATCH * M1C * 4);
  float* pos1  = (float*)alloc((size_t)BATCH * M1C * 3 * 4);
  int*   nidx1 = (int*)  alloc((size_t)BATCH * M1C * KNBR * 4);
  float* nd21  = (float*)alloc((size_t)BATCH * M1C * KNBR * 4);
  f16*   x1    = (f16*)  alloc((size_t)BATCH * M1C * 128 * 2);
  int*   idx2  = (int*)  alloc((size_t)BATCH * M2C * 4);
  float* pos2  = (float*)alloc((size_t)BATCH * M2C * 3 * 4);
  int*   nidx2 = (int*)  alloc((size_t)BATCH * M2C * KNBR * 4);
  float* nd22  = (float*)alloc((size_t)BATCH * M2C * KNBR * 4);
  f16*   x2    = (f16*)  alloc((size_t)BATCH * M2C * 256 * 2);
  f16*   sa3in = (f16*)  alloc((size_t)BATCH * M2C * 288 * 2);
  f16*   x3    = (f16*)  alloc((size_t)BATCH * 1024 * 2);
  f16*   f3    = (f16*)  alloc((size_t)BATCH * M2C * 256 * 2);
  f16*   f2    = (f16*)  alloc((size_t)BATCH * M1C * 128 * 2);
  int*   iidx2 = (int*)  alloc((size_t)BATCH * M1C * 3 * 4);
  float* id2_2 = (float*)alloc((size_t)BATCH * M1C * 3 * 4);
  int*   iidx1 = (int*)  alloc((size_t)BATCH * NPTS * 3 * 4);
  float* id2_1 = (float*)alloc((size_t)BATCH * NPTS * 3 * 4);
  const size_t GELEMS = (size_t)BATCH * NPTS * 160;  // largest staged matrix
  f16* GA = (f16*)alloc(GELEMS * 2);
  f16* GB = (f16*)alloc(GELEMS * 2);
  f16* wp[18];
  for (int i = 0; i < 18; ++i)
    wp[i] = (f16*)alloc((size_t)wlist[i].cinpad * wlist[i].cout * 2);

  auto grid1 = [](int total) { return (total + 255) / 256; };

  // weights f32 -> padded fragment-major f16
  for (int i = 0; i < 18; ++i) {
    int total = wlist[i].cinpad * wlist[i].cout;
    pack_w_frag<<<grid1(total), 256, 0, stream>>>(P[wlist[i].pidx], wlist[i].cin,
                                                  wlist[i].cout, wlist[i].cinpad,
                                                  wp[i]);
  }
  copy_f32_mat<<<grid1(BATCH * NPTS * FIN), 256, 0, stream>>>(
      x, BATCH * NPTS, FIN, xh, FIN, 0);

  // ---- SA1 ----
  fps_kernel<<<BATCH, 256, 0, stream>>>(pos, NPTS, M1C, idx1);
  gather_pos<<<grid1(BATCH * M1C * 3), 256, 0, stream>>>(pos, idx1, NPTS, M1C, pos1);
  knn_kernel<<<BATCH * M1C, 256, 0, stream>>>(pos1, pos, M1C, NPTS, KNBR, nidx1, nd21);
  {
    size_t sh = (size_t)(2 * 256 * 128 + 64 * 128) * sizeof(f16);
    sa_fused<4, 128, 8192><<<BATCH * ((M1C + 3) / 4), 256, sh, stream>>>(
        xh, FIN, NPTS, pos, pos1, nidx1, nd21, 0.04f, M1C,
        wp[0], P[1], 32, 64, wp[1], P[3], 64, wp[2], P[5], 128, x1);
  }

  // ---- SA2 ----
  fps_kernel<<<BATCH, 256, 0, stream>>>(pos1, M1C, M2C, idx2);
  gather_pos<<<grid1(BATCH * M2C * 3), 256, 0, stream>>>(pos1, idx2, M1C, M2C, pos2);
  knn_kernel<<<BATCH * M2C, 256, 0, stream>>>(pos2, pos1, M2C, M1C, KNBR, nidx2, nd22);
  {
    size_t sh = (size_t)(2 * 128 * 256 + 128 * 256) * sizeof(f16);
    sa_fused<2, 256, 32768><<<BATCH * ((M2C + 1) / 2), 256, sh, stream>>>(
        x1, 128, M1C, pos1, pos2, nidx2, nd22, 0.16f, M2C,
        wp[3], P[7], 160, 128, wp[4], P[9], 128, wp[5], P[11], 256, x2);
  }

  auto gemm = [&](const f16* A, int wi, const float* bias, f16* Out, int rows,
                  int K, int Cout, int act) {
    int tiles = ((rows + 15) / 16) * (Cout / 16);
    gemm_f16<<<(tiles + 7) / 8, 256, 0, stream>>>(A, wp[wi], bias, Out, rows, K,
                                                  Cout, act);
  };

  // ---- SA3 (global) ----
  fill_f16<<<grid1(BATCH * M2C * 288), 256, 0, stream>>>(sa3in, BATCH * M2C * 288);
  copy_f16_mat<<<grid1(BATCH * M2C * 256), 256, 0, stream>>>(
      x2, BATCH * M2C, 256, sa3in, 288, 0);
  copy_f32_mat<<<grid1(BATCH * M2C * 3), 256, 0, stream>>>(
      pos2, BATCH * M2C, 3, sa3in, 288, 256);
  gemm(sa3in, 6, P[13], GA, BATCH * M2C, 288, 256, 2);
  gemm(GA, 7, P[15], GB, BATCH * M2C, 256, 512, 2);
  gemm(GB, 8, P[17], GA, BATCH * M2C, 512, 1024, 2);
  rowmax_kernel<<<grid1(BATCH * 1024), 256, 0, stream>>>(GA, M2C, 1024, x3);

  // ---- FP3 (coarse point is global -> broadcast) ----
  bcast_x3<<<grid1(BATCH * M2C * 1024), 256, 0, stream>>>(x3, GB, M2C, 1280);
  copy_f16_mat<<<grid1(BATCH * M2C * 256), 256, 0, stream>>>(
      x2, BATCH * M2C, 256, GB, 1280, 1024);
  gemm(GB, 9, P[19], GA, BATCH * M2C, 1280, 256, 2);
  gemm(GA, 10, P[21], f3, BATCH * M2C, 256, 256, 2);

  // ---- FP2 ----
  knn_kernel<<<BATCH * M1C, 256, 0, stream>>>(pos1, pos2, M1C, M2C, 3, iidx2, id2_2);
  interp_kernel<<<grid1(BATCH * M1C * 256), 256, 0, stream>>>(
      f3, iidx2, id2_2, M1C, M2C, 256, 3, GA, 384, 0);
  copy_f16_mat<<<grid1(BATCH * M1C * 128), 256, 0, stream>>>(
      x1, BATCH * M1C, 128, GA, 384, 256);
  gemm(GA, 11, P[23], GB, BATCH * M1C, 384, 256, 2);
  gemm(GB, 12, P[25], f2, BATCH * M1C, 256, 128, 2);

  // ---- FP1 ----
  knn_kernel<<<BATCH * NPTS, 256, 0, stream>>>(pos, pos1, NPTS, M1C, 3, iidx1, id2_1);
  fill_f16<<<grid1((int)GELEMS), 256, 0, stream>>>(GA, (int)GELEMS);
  interp_kernel<<<grid1(BATCH * NPTS * 128), 256, 0, stream>>>(
      f2, iidx1, id2_1, NPTS, M1C, 128, 3, GA, 160, 0);
  copy_f32_mat<<<grid1(BATCH * NPTS * FIN), 256, 0, stream>>>(
      x, BATCH * NPTS, FIN, GA, 160, 128);
  gemm(GA, 13, P[27], GB, BATCH * NPTS, 160, 128, 2);
  gemm(GB, 14, P[29], GA, BATCH * NPTS, 128, 128, 2);
  gemm(GA, 15, P[31], GB, BATCH * NPTS, 128, 128, 2);

  // ---- head + log_softmax ----
  gemm(GB, 16, P[33], GA, BATCH * NPTS, 128, 128, 1);  // relu, no BN
  gemm(GA, 17, P[35], GB, BATCH * NPTS, 128, 64, 0);   // linear
  head_final<<<grid1(BATCH * NPTS), 256, 0, stream>>>(GB, P[36], P[37],
                                                      (float*)d_out, BATCH * NPTS);
}
